// Block_472446402621
// MI455X (gfx1250) — matrix operand order repaired and checked by reading, not compiled, not run on hardware
//
#include <hip/hip_runtime.h>
#include <hip/hip_bf16.h>
#include <math.h>

typedef __attribute__((ext_vector_type(16))) _Float16 v16h;
typedef __attribute__((ext_vector_type(8)))  float    v8f;

#define D_MODEL 1024
#define D_STATE 16
#define D_CONV  8
#define D_INNER 1024
#define DT_RANK 64
#define FFN_HID 4096
#define BATCH   2
#define SEQ     4096
#define ROWS    (BATCH * SEQ)   // 8192 token rows

// ---------------------------------------------------------------------------
// Elementwise f32 -> f16 conversion (weights + misc)
// ---------------------------------------------------------------------------
__global__ void cvt_f32_to_f16(const float* __restrict__ src,
                               _Float16* __restrict__ dst, int n) {
    int i = blockIdx.x * blockDim.x + threadIdx.x;
    if (i < n) dst[i] = (_Float16)src[i];
}

// ---------------------------------------------------------------------------
// LayerNorm over last dim (1024), fp32 in -> f16 out. One block per row.
// ---------------------------------------------------------------------------
__global__ void layernorm_f16(const float* __restrict__ x,
                              const float* __restrict__ g,
                              const float* __restrict__ b,
                              _Float16* __restrict__ out) {
    __shared__ float s_sum[256];
    __shared__ float s_sq[256];
    const int row = blockIdx.x;
    const int tid = threadIdx.x;
    const float* xr = x + (size_t)row * D_MODEL;

    float sum = 0.f, sq = 0.f;
    for (int i = tid; i < D_MODEL; i += 256) {
        float v = xr[i];
        sum += v; sq += v * v;
    }
    s_sum[tid] = sum; s_sq[tid] = sq;
    __syncthreads();
    for (int s = 128; s > 0; s >>= 1) {
        if (tid < s) { s_sum[tid] += s_sum[tid + s]; s_sq[tid] += s_sq[tid + s]; }
        __syncthreads();
    }
    const float mu  = s_sum[0] * (1.0f / D_MODEL);
    const float var = s_sq[0] * (1.0f / D_MODEL) - mu * mu;
    const float inv = rsqrtf(var + 1e-5f);

    _Float16* orow = out + (size_t)row * D_MODEL;
    for (int i = tid; i < D_MODEL; i += 256) {
        orow[i] = (_Float16)((xr[i] - mu) * inv * g[i] + b[i]);
    }
}

// ---------------------------------------------------------------------------
// WMMA GEMM: out[M,N] = A[M,K] (f16, row-major) x W[N,K]^T (f16, row-major)
// Register-blocked: each wave computes a 16 x (16*NT) strip. The A fragment
// is loaded once per 32-deep K step and reused across NT V_WMMA_F32_16X16X32_F16
// issues (raises L2 arithmetic intensity ~2x vs a single-tile wave and gives
// the scheduler NT back-to-back WMMAs per VMEM burst).
// Fragment packing uses contiguous K order (k = 16*half + s) for both A and B.
// ---------------------------------------------------------------------------
#define EPI_STORE_F16      0
#define EPI_STORE_F32      1
#define EPI_BIAS_RELU_F16  2
#define EPI_BIAS_SP_F32    3   // bias + softplus -> f32
#define EPI_ADDRES_F32     4   // out = gemm + res
#define EPI_BIAS_ADDRES    5   // out = res + gemm + bias (res may alias out)

union FragU { v16h h; uint32_t u[8]; };

template <int NT>
__global__ void gemm_wmma_f16(const _Float16* __restrict__ A,
                              const _Float16* __restrict__ W,
                              float* __restrict__ outF,
                              _Float16* __restrict__ outH,
                              const float* __restrict__ bias,
                              const float* __restrict__ res,
                              int M, int N, int K, int mode) {
    const int wavesPerBlock = blockDim.x >> 5;
    const int waveId  = blockIdx.x * wavesPerBlock + (threadIdx.x >> 5);
    const int lane    = threadIdx.x & 31;
    const int tilesN  = N >> 4;
    const int tilesM  = M >> 4;
    const int groupsN = tilesN / NT;
    const int tm = waveId / groupsN;
    const int tg = waveId % groupsN;
    if (tm >= tilesM) return;                 // wave-uniform exit (EXEC stays all-1)

    const int row0 = tm << 4;
    const int col0 = tg * (NT << 4);
    const int m       = lane & 15;
    const int half_id = lane >> 4;

    v8f acc[NT];
    #pragma unroll
    for (int j = 0; j < NT; ++j) { v8f z = {}; acc[j] = z; }

    const _Float16* Abase = A + (size_t)(row0 + m) * K + half_id * 16;
    const _Float16* Wbase[NT];
    #pragma unroll
    for (int j = 0; j < NT; ++j)
        Wbase[j] = W + (size_t)(col0 + (j << 4) + m) * K + half_id * 16;

    for (int k0 = 0; k0 < K; k0 += 32) {
        // A fragment: lanes 0-15 row M=m hold K = k0+0..15;
        //             lanes 16-31 hold K = k0+16..31.
        FragU a;
        const uint32_t* Ap = (const uint32_t*)(Abase + k0);
        #pragma unroll
        for (int i = 0; i < 4; ++i) a.u[i]     = Ap[i];      // K = off+2i,2i+1
        #pragma unroll
        for (int i = 0; i < 4; ++i) a.u[4 + i] = Ap[4 + i];  // K = off+8+2i,...
        __builtin_prefetch(Abase + k0 + 32, 0, 0);           // global_prefetch_b8

        // B fragments: column n=m; lanes 0-15 hold K = k0+0..15 (2/VGPR),
        //              lanes 16-31 hold K = k0+16..31.
        FragU b[NT];
        #pragma unroll
        for (int j = 0; j < NT; ++j) {
            const uint32_t* Wp = (const uint32_t*)(Wbase[j] + k0);
            #pragma unroll
            for (int i = 0; i < 8; ++i) b[j].u[i] = Wp[i];
        }
        #pragma unroll
        for (int j = 0; j < NT; ++j) {
            acc[j] = __builtin_amdgcn_wmma_f32_16x16x32_f16(
                false, a.h, false, b[j].h, (short)0, acc[j], false, false);
        }
    }

    // C/D layout: VGPR r -> M = row0 + r + 8*half_id, N = col + m
    #pragma unroll
    for (int j = 0; j < NT; ++j) {
        const int col = col0 + (j << 4) + m;
        #pragma unroll
        for (int r = 0; r < 8; ++r) {
            const int orow = row0 + r + (half_id << 3);
            const size_t idx = (size_t)orow * N + col;
            float v = acc[j][r];
            switch (mode) {
                case EPI_STORE_F16:
                    outH[idx] = (_Float16)v; break;
                case EPI_STORE_F32:
                    outF[idx] = v; break;
                case EPI_BIAS_RELU_F16: {
                    float t = v + bias[col];
                    outH[idx] = (_Float16)(t > 0.f ? t : 0.f); break;
                }
                case EPI_BIAS_SP_F32: {
                    float t = v + bias[col];
                    outF[idx] = (t > 20.f) ? t : log1pf(__expf(t)); break;
                }
                case EPI_ADDRES_F32:
                    outF[idx] = v + res[idx]; break;
                case EPI_BIAS_ADDRES:
                    outF[idx] = res[idx] + v + bias[col]; break;
            }
        }
    }
}

// ---------------------------------------------------------------------------
// Depthwise causal conv (width 8) + SiLU.
// xin = first D_INNER columns of xz16 (row stride 2*D_INNER).
// ---------------------------------------------------------------------------
__global__ void conv_silu(const _Float16* __restrict__ xz16,
                          const float* __restrict__ conv_w,
                          const float* __restrict__ conv_b,
                          _Float16* __restrict__ xc16) {
    int idx = blockIdx.x * blockDim.x + threadIdx.x;
    if (idx >= ROWS * D_INNER) return;
    const int d   = idx & (D_INNER - 1);
    const int row = idx >> 10;               // b*SEQ + t
    const int t   = row & (SEQ - 1);
    const int rb0 = row - t;                 // start row of this batch

    float acc = conv_b[d];
    #pragma unroll
    for (int k = 0; k < D_CONV; ++k) {
        const int tt = t - (D_CONV - 1) + k;
        if (tt >= 0) {
            float xin = (float)xz16[(size_t)(rb0 + tt) * (2 * D_INNER) + d];
            acc += xin * conv_w[d * D_CONV + k];
        }
    }
    const float s = acc / (1.f + __expf(-acc));   // silu
    xc16[idx] = (_Float16)s;
}

// ---------------------------------------------------------------------------
// Extract dt (first DT_RANK cols of dbl) as f16 for the dt_proj GEMM.
// ---------------------------------------------------------------------------
__global__ void split_dt(const float* __restrict__ dbl,
                         _Float16* __restrict__ dt16) {
    int idx = blockIdx.x * blockDim.x + threadIdx.x;
    if (idx >= ROWS * DT_RANK) return;
    const int row = idx / DT_RANK;
    const int r   = idx - row * DT_RANK;
    dt16[idx] = (_Float16)dbl[(size_t)row * (DT_RANK + 2 * D_STATE) + r];
}

// ---------------------------------------------------------------------------
// Selective scan, fused with skip (D_param) and y*silu(z) gate.
// Grid: BATCH * (D_INNER/256) blocks; one channel d per thread,
// 16-state recurrence in registers, B/C broadcast through LDS per step.
// ---------------------------------------------------------------------------
__global__ void selective_scan(const float* __restrict__ delta,
                               const _Float16* __restrict__ xc16,
                               const float* __restrict__ dbl,
                               const _Float16* __restrict__ xz16,
                               const float* __restrict__ A_log,
                               const float* __restrict__ D_param,
                               _Float16* __restrict__ y16) {
    const int chunks = D_INNER / 256;
    const int b  = blockIdx.x / chunks;
    const int d  = (blockIdx.x % chunks) * 256 + threadIdx.x;

    float Aneg[D_STATE];
    float h[D_STATE];
    #pragma unroll
    for (int n = 0; n < D_STATE; ++n) {
        Aneg[n] = -__expf(A_log[d * D_STATE + n]);
        h[n] = 0.f;
    }
    const float Dp = D_param[d];

    __shared__ float sBC[2 * D_STATE];

    for (int t = 0; t < SEQ; ++t) {
        const int row = b * SEQ + t;
        if (threadIdx.x < 2 * D_STATE)
            sBC[threadIdx.x] =
                dbl[(size_t)row * (DT_RANK + 2 * D_STATE) + DT_RANK + threadIdx.x];
        __syncthreads();

        const float dl = delta[(size_t)row * D_INNER + d];
        const float xc = (float)xc16[(size_t)row * D_INNER + d];
        float y = 0.f;
        #pragma unroll
        for (int n = 0; n < D_STATE; ++n) {
            const float dA = __expf(dl * Aneg[n]);
            h[n] = h[n] * dA + dl * sBC[n] * xc;
            y += h[n] * sBC[D_STATE + n];
        }
        y += xc * Dp;

        const float z  = (float)xz16[(size_t)row * (2 * D_INNER) + D_INNER + d];
        const float sz = z / (1.f + __expf(-z));
        y16[(size_t)row * D_INNER + d] = (_Float16)(y * sz);
        __syncthreads();
    }
}

// ---------------------------------------------------------------------------
// Host-side launch
// ---------------------------------------------------------------------------
static inline void launch_gemm(const _Float16* A, const _Float16* W,
                               float* outF, _Float16* outH,
                               const float* bias, const float* res,
                               int M, int N, int K, int mode, hipStream_t s) {
    const int tilesM = M >> 4;
    const int tilesN = N >> 4;
    if ((tilesN & 3) == 0) {
        const int waves  = tilesM * (tilesN >> 2);
        const int blocks = (waves + 7) / 8;      // 8 waves / 256-thread block
        gemm_wmma_f16<4><<<blocks, 256, 0, s>>>(A, W, outF, outH, bias, res,
                                                M, N, K, mode);
    } else {
        const int waves  = tilesM * (tilesN >> 1);
        const int blocks = (waves + 7) / 8;
        gemm_wmma_f16<2><<<blocks, 256, 0, s>>>(A, W, outF, outH, bias, res,
                                                M, N, K, mode);
    }
}

extern "C" void kernel_launch(void* const* d_in, const int* in_sizes, int n_in,
                              void* d_out, int out_size, void* d_ws, size_t ws_size,
                              hipStream_t stream) {
    const float* x          = (const float*)d_in[0];
    const float* in_proj_w  = (const float*)d_in[1];
    const float* conv_w     = (const float*)d_in[2];
    const float* conv_b     = (const float*)d_in[3];
    const float* x_proj_w   = (const float*)d_in[4];
    const float* dt_proj_w  = (const float*)d_in[5];
    const float* dt_proj_b  = (const float*)d_in[6];
    const float* A_log      = (const float*)d_in[7];
    const float* D_param    = (const float*)d_in[8];
    const float* out_proj_w = (const float*)d_in[9];
    const float* ln1_g      = (const float*)d_in[10];
    const float* ln1_b      = (const float*)d_in[11];
    const float* ln2_g      = (const float*)d_in[12];
    const float* ln2_b      = (const float*)d_in[13];
    const float* w1         = (const float*)d_in[14];
    const float* b1         = (const float*)d_in[15];
    const float* w2         = (const float*)d_in[16];
    const float* b2         = (const float*)d_in[17];
    float* out = (float*)d_out;

    char* ws = (char*)d_ws;
    const size_t MB = 1024ull * 1024ull;
    // phase-A buffers
    _Float16* h16      = (_Float16*)(ws + 0);          // 16 MB (reused as h2 in phase B)
    _Float16* xz16     = (_Float16*)(ws + 16 * MB);    // 32 MB
    _Float16* xc16     = (_Float16*)(ws + 48 * MB);    // 16 MB
    float*    dbl      = (float*)   (ws + 64 * MB);    //  3 MB (dt|B|C, 96 cols)
    _Float16* dt16     = (_Float16*)(ws + 68 * MB);    //  1 MB
    float*    delta    = (float*)   (ws + 69 * MB);    // 32 MB
    // f16 weights
    _Float16* in_w16   = (_Float16*)(ws + 101 * MB);   //  4 MB
    _Float16* xp_w16   = (_Float16*)(ws + 105 * MB);   //  0.19 MB
    _Float16* dt_w16   = (_Float16*)(ws + 106 * MB);   //  0.13 MB
    _Float16* op_w16   = (_Float16*)(ws + 107 * MB);   //  2 MB
    _Float16* w1_16    = (_Float16*)(ws + 109 * MB);   //  8 MB
    _Float16* w2_16    = (_Float16*)(ws + 117 * MB);   //  8 MB
    _Float16* y16      = (_Float16*)(ws + 125 * MB);   // 16 MB
    // phase-B buffer reusing dead phase-A region [16MB, 80MB)
    _Float16* ffnh16   = (_Float16*)(ws + 16 * MB);    // 64 MB

    auto cvt = [&](const float* src, _Float16* dst, int n) {
        cvt_f32_to_f16<<<(n + 255) / 256, 256, 0, stream>>>(src, dst, n);
    };

    // --- weight conversions (f32 -> f16) ---
    cvt(in_proj_w,  in_w16, 2 * D_INNER * D_MODEL);
    cvt(x_proj_w,   xp_w16, (DT_RANK + 2 * D_STATE) * D_INNER);
    cvt(dt_proj_w,  dt_w16, D_INNER * DT_RANK);
    cvt(out_proj_w, op_w16, D_MODEL * D_INNER);
    cvt(w1,         w1_16,  FFN_HID * D_MODEL);
    cvt(w2,         w2_16,  D_MODEL * FFN_HID);

    // --- Mamba branch ---
    layernorm_f16<<<ROWS, 256, 0, stream>>>(x, ln1_g, ln1_b, h16);

    // xz = LN(x) @ in_proj_w^T              [8192 x 2048]
    launch_gemm(h16, in_w16, nullptr, xz16, nullptr, nullptr,
                ROWS, 2 * D_INNER, D_MODEL, EPI_STORE_F16, stream);

    // xc = silu(causal_conv8(xin) + b)      [8192 x 1024]
    conv_silu<<<(ROWS * D_INNER + 255) / 256, 256, 0, stream>>>(
        xz16, conv_w, conv_b, xc16);

    // dbl = xc @ x_proj_w^T                 [8192 x 96]
    launch_gemm(xc16, xp_w16, dbl, nullptr, nullptr, nullptr,
                ROWS, DT_RANK + 2 * D_STATE, D_INNER, EPI_STORE_F32, stream);

    split_dt<<<(ROWS * DT_RANK + 255) / 256, 256, 0, stream>>>(dbl, dt16);

    // delta = softplus(dt @ dt_proj_w^T + dt_b)   [8192 x 1024]
    launch_gemm(dt16, dt_w16, delta, nullptr, dt_proj_b, nullptr,
                ROWS, D_INNER, DT_RANK, EPI_BIAS_SP_F32, stream);

    // y16 = gate(scan(...))                 [8192 x 1024]
    selective_scan<<<BATCH * (D_INNER / 256), 256, 0, stream>>>(
        delta, xc16, dbl, xz16, A_log, D_param, y16);

    // x2 = x + y @ out_proj_w^T  -> written to d_out
    launch_gemm(y16, op_w16, out, nullptr, nullptr, x,
                ROWS, D_MODEL, D_INNER, EPI_ADDRES_F32, stream);

    // --- FFN branch ---
    layernorm_f16<<<ROWS, 256, 0, stream>>>(out, ln2_g, ln2_b, h16);

    // h = relu(LN(x2) @ w1^T + b1)          [8192 x 4096]
    launch_gemm(h16, w1_16, nullptr, ffnh16, b1, nullptr,
                ROWS, FFN_HID, D_MODEL, EPI_BIAS_RELU_F16, stream);

    // out = x2 + h @ w2^T + b2  (in-place residual on d_out)
    launch_gemm(ffnh16, w2_16, out, nullptr, b2, out,
                ROWS, D_MODEL, FFN_HID, EPI_BIAS_ADDRES, stream);
}
